// LinkPred_23106924052715
// MI455X (gfx1250) — matrix operand order repaired and checked by reading, not compiled, not run on hardware
//
#include <hip/hip_runtime.h>
#include <hip/hip_bf16.h>
#include <cstdint>
#include <cstddef>

// ---------------- problem constants (match reference) ----------------
#define N_NODES 50000
#define N_EDGES 800000
#define NFEAT   512
#define NHID    256
#define NCLASS  128
#define N_UV    2048

#define USE_ASYNC_LDS 1   // CDNA5 global_load_async_to_lds for SpMM edge staging

// ---------------- CDNA5 WMMA types ----------------
typedef __attribute__((ext_vector_type(16))) __bf16 v16bf;
typedef __attribute__((ext_vector_type(8)))  float  v8f;

union FragAB { v16bf v; uint4 q[2]; };
union FragC  { v8f   v; float f[8]; };

// round-to-nearest-even f32 -> bf16 (bit pattern as ushort)
static __device__ __forceinline__ unsigned short f2bf(float f) {
  union { float f; unsigned u; } c; c.f = f;
  unsigned r = 0x7FFFu + ((c.u >> 16) & 1u);
  return (unsigned short)((c.u + r) >> 16);
}

// ---------------- elementwise helpers ----------------
__global__ void k_f32_to_bf16(const float* __restrict__ in,
                              unsigned short* __restrict__ out, long n) {
  long i = (long)blockIdx.x * blockDim.x + threadIdx.x;
  if (i < n) out[i] = f2bf(in[i]);
}

__global__ void k_zero_f32(float* __restrict__ p, long n) {
  long i = (long)blockIdx.x * blockDim.x + threadIdx.x;
  if (i < n) p[i] = 0.0f;
}

// out[c*rows + r] = bf16(in[r*cols + c])   (W -> W^T in bf16)
__global__ void k_transpose_bf16(const float* __restrict__ in,
                                 unsigned short* __restrict__ out,
                                 int rows, int cols) {
  int c = blockIdx.x * blockDim.x + threadIdx.x;
  int r = blockIdx.y;
  if (c < cols) out[(long)c * rows + r] = f2bf(in[(long)r * cols + c]);
}

// hb = bf16(relu(acc + b[f]))   (fmask = F-1, F power of two)
__global__ void k_bias_relu_bf16(const float* __restrict__ acc,
                                 const float* __restrict__ b,
                                 unsigned short* __restrict__ out,
                                 long n, int fmask) {
  long i = (long)blockIdx.x * blockDim.x + threadIdx.x;
  if (i < n) {
    float v = acc[i] + b[i & fmask];
    out[i] = f2bf(v > 0.0f ? v : 0.0f);
  }
}

// out[i][f] = bf16(acc[idx[i]][f] + b[f])
__global__ void k_gather_bias_bf16(const float* __restrict__ acc,
                                   const float* __restrict__ b,
                                   const int* __restrict__ idx,
                                   unsigned short* __restrict__ out, int F) {
  int i = blockIdx.x;
  int f = threadIdx.x;
  out[(long)i * F + f] = f2bf(acc[(long)idx[i] * F + f] + b[f]);
}

// ---------------- COO SpMM: acc[row] += val * S[col] ----------------
// blockDim.x == F. Edge tiles (128 triples) are staged into LDS once per
// block via CDNA5 async global->LDS copies (ASYNCcnt), then all F threads
// consume them as LDS broadcasts. Sorted adj_row -> run-length accumulate
// in registers; one f32 atomic per (row-run, feature). S/acc are
// L2-resident on MI455X (51MB each vs 192MB L2).
#define ET  128   // edges per LDS tile
#define EPB 512   // edges per block

struct EdgeTile { int r[ET]; int c[ET]; float v[ET]; };

__global__ void k_spmm(const int* __restrict__ rows,
                       const int* __restrict__ cols,
                       const float* __restrict__ vals,
                       const float* __restrict__ S,
                       float* __restrict__ acc,
                       long E, int F) {
  __shared__ EdgeTile sm;
  const int f = threadIdx.x;
  long e0 = (long)blockIdx.x * EPB;
  long e1 = e0 + EPB; if (e1 > E) e1 = E;

  float a = 0.0f; int cur = -1;
  for (long t0 = e0; t0 < e1; t0 += ET) {
    int cnt = (int)((e1 - t0) < ET ? (e1 - t0) : ET);
    __syncthreads();                       // previous tile fully consumed
    if (f < cnt) {
#if USE_ASYNC_LDS
      unsigned lo_r = (unsigned)(unsigned long long)&sm.r[f];
      unsigned lo_c = (unsigned)(unsigned long long)&sm.c[f];
      unsigned lo_v = (unsigned)(unsigned long long)&sm.v[f];
      unsigned long long ga_r = (unsigned long long)(rows + t0 + f);
      unsigned long long ga_c = (unsigned long long)(cols + t0 + f);
      unsigned long long ga_v = (unsigned long long)(vals + t0 + f);
      asm volatile("global_load_async_to_lds_b32 %0, %1, off"
                   :: "v"(lo_r), "v"(ga_r) : "memory");
      asm volatile("global_load_async_to_lds_b32 %0, %1, off"
                   :: "v"(lo_c), "v"(ga_c) : "memory");
      asm volatile("global_load_async_to_lds_b32 %0, %1, off"
                   :: "v"(lo_v), "v"(ga_v) : "memory");
#else
      sm.r[f] = rows[t0 + f];
      sm.c[f] = cols[t0 + f];
      sm.v[f] = vals[t0 + f];
#endif
    }
#if USE_ASYNC_LDS
    asm volatile("s_wait_asynccnt 0x0" ::: "memory");
#endif
    __syncthreads();
    for (int j = 0; j < cnt; ++j) {
      int r = sm.r[j]; int c = sm.c[j]; float w = sm.v[j];
      if (r != cur) {
        if (cur >= 0) atomicAdd(&acc[(long)cur * F + f], a);
        cur = r; a = 0.0f;
      }
      a = fmaf(w, S[(long)c * F + f], a);
    }
  }
  if (cur >= 0) atomicAdd(&acc[(long)cur * F + f], a);
}

// ---------------- bf16 WMMA GEMM (B supplied transposed) ----------------
// C[M,N](f32) = A[M,K](bf16 rowmajor) * BT[N,K](bf16 rowmajor)^T
// No LDS, no barriers: a WMMA B-fragment is a K-contiguous read of BT,
// identical addressing to the A-fragment. BT matrices are small/cache
// resident, so L0/L2 provides the reuse that LDS staging would.
// block = 256 threads = 8 waves; wave tile = 32(M) x 64(N); K step 32
// -> 8 v_wmma per 12 b128 loads. mode: 0 f32, 1 f32+sigmoid, 2 bf16.
__launch_bounds__(256)
__global__ void k_gemm_bf16(const unsigned short* __restrict__ A,
                            const unsigned short* __restrict__ BT,
                            float* __restrict__ Cf,
                            unsigned short* __restrict__ Cb,
                            int M, int N, int K, int mode) {
  const int tid  = threadIdx.x;
  const int lane = tid & 31;
  const int wave = tid >> 5;
  const int half = (lane >> 4) & 1;     // lane group 0..15 vs 16..31
  const int lr   = lane & 15;
  const int n0   = blockIdx.x * 64;
  const int m0   = blockIdx.y * 256 + wave * 32;
  if (m0 >= M) return;                  // no barriers -> safe early exit
  const bool act1 = (m0 + 16) < M;      // M % 16 == 0 for all shapes

  FragC acc[2][4];
  #pragma unroll
  for (int h = 0; h < 2; ++h)
    #pragma unroll
    for (int t = 0; t < 4; ++t)
      #pragma unroll
      for (int i = 0; i < 8; ++i) acc[h][t].f[i] = 0.0f;

  // A frag (16x32): lanes0-15 K +0..7/+16..23, lanes16-31 +8..15/+24..31
  const unsigned short* a0p = A + (long)(m0 + lr) * K + half * 8;
  const unsigned short* a1p = act1 ? (A + (long)(m0 + 16 + lr) * K + half * 8) : a0p;
  // B frag (32x16): lanes0-15 K +0..7/+16..23, lanes16-31 +8..15/+24..31
  const unsigned short* bp0 = BT + (long)(n0 + lr) * K + half * 8;

  for (int kb = 0; kb < K; kb += 32) {
    FragAB a0, a1, b[4];
    a0.q[0] = *(const uint4*)(a0p + kb);
    a0.q[1] = *(const uint4*)(a0p + kb + 16);
    a1.q[0] = *(const uint4*)(a1p + kb);
    a1.q[1] = *(const uint4*)(a1p + kb + 16);
    #pragma unroll
    for (int t = 0; t < 4; ++t) {
      const unsigned short* bp = bp0 + (long)t * 16 * K + kb;
      b[t].q[0] = *(const uint4*)(bp);
      b[t].q[1] = *(const uint4*)(bp + 16);
    }
    if (kb + 32 < K) {
      __builtin_prefetch(a0p + kb + 32, 0, 0);
      __builtin_prefetch(bp0 + kb + 32, 0, 0);
    }
    #pragma unroll
    for (int t = 0; t < 4; ++t) {
      acc[0][t].v = __builtin_amdgcn_wmma_f32_16x16x32_bf16(
          false, a0.v, false, b[t].v, (short)0, acc[0][t].v, false, false);
      acc[1][t].v = __builtin_amdgcn_wmma_f32_16x16x32_bf16(
          false, a1.v, false, b[t].v, (short)0, acc[1][t].v, false, false);
    }
  }

  #pragma unroll
  for (int h = 0; h < 2; ++h) {
    if (h == 1 && !act1) break;
    #pragma unroll
    for (int t = 0; t < 4; ++t) {
      #pragma unroll
      for (int i = 0; i < 8; ++i) {
        int row = m0 + h * 16 + i + half * 8;   // C layout: VGPR i -> M
        int col = n0 + t * 16 + lr;             // N = lane & 15
        float vv = acc[h][t].f[i];
        if (mode == 1) vv = __builtin_amdgcn_rcpf(1.0f + __expf(-vv));
        if (mode == 2) Cb[(long)row * N + col] = f2bf(vv);
        else           Cf[(long)row * N + col] = vv;
      }
    }
  }
}

// ---------------- host orchestration ----------------
extern "C" void kernel_launch(void* const* d_in, const int* in_sizes, int n_in,
                              void* d_out, int out_size, void* d_ws, size_t ws_size,
                              hipStream_t stream) {
  const int*   u       = (const int*)  d_in[0];
  const int*   v       = (const int*)  d_in[1];
  const float* x       = (const float*)d_in[2];
  const int*   adj_row = (const int*)  d_in[3];
  const int*   adj_col = (const int*)  d_in[4];
  const float* adj_val = (const float*)d_in[5];
  const float* W1      = (const float*)d_in[6];
  const float* b1      = (const float*)d_in[7];
  const float* W2      = (const float*)d_in[8];
  const float* b2      = (const float*)d_in[9];
  const float* We      = (const float*)d_in[10];
  float*       out     = (float*)d_out;

  // workspace carve-up (stream-ordered aliasing noted)
  char* ws = (char*)d_ws;
  size_t off = 0;
  auto carve = [&](size_t bytes) -> char* {
    char* p = ws + off; off += (bytes + 255) & ~(size_t)255; return p;
  };
  unsigned short* xb   = (unsigned short*)carve((size_t)N_NODES * NFEAT * 2); // 51.2MB
  float*          s1   = (float*)         carve((size_t)N_NODES * NHID  * 4); // 51.2MB
  unsigned short* hb   = (unsigned short*)carve((size_t)N_NODES * NHID  * 2); // 25.6MB
  unsigned short* W1T  = (unsigned short*)carve((size_t)NHID * NFEAT * 2);    // [N][K]
  unsigned short* W2T  = (unsigned short*)carve((size_t)NCLASS * NHID * 2);   // [N][K]
  unsigned short* Web  = (unsigned short*)carve((size_t)NCLASS * NCLASS * 2); // We as-is == (We^T)^T
  unsigned short* zuin = (unsigned short*)carve((size_t)N_UV * NCLASS * 2);
  unsigned short* zvb  = (unsigned short*)carve((size_t)N_UV * NCLASS * 2);   // == BT of GEMM4
  unsigned short* zub  = (unsigned short*)carve((size_t)N_UV * NCLASS * 2);
  // aliases (safe: consumers of old contents run earlier on the stream)
  float* acc1 = (float*)xb;   // xb dead after GEMM1
  float* s2   = s1;           // s1 dead after SpMM1
  float* acc2 = (float*)hb;   // hb dead after GEMM2

  const long nx  = (long)N_NODES * NFEAT;
  const long nh1 = (long)N_NODES * NHID;
  const long nh2 = (long)N_NODES * NCLASS;

  // 1. precision conversions / BT layouts
  k_f32_to_bf16<<<(unsigned)((nx + 255) / 256), 256, 0, stream>>>(x, xb, nx);
  k_transpose_bf16<<<dim3((NHID + 127) / 128, NFEAT), 128, 0, stream>>>(W1, W1T, NFEAT, NHID);
  k_transpose_bf16<<<dim3((NCLASS + 127) / 128, NHID), 128, 0, stream>>>(W2, W2T, NHID, NCLASS);
  k_f32_to_bf16<<<(NCLASS * NCLASS + 255) / 256, 256, 0, stream>>>(We, Web, NCLASS * NCLASS);

  // 2. layer 1 GEMM: s1 = xb @ W1   (50000x512x256, bf16 WMMA)
  k_gemm_bf16<<<dim3(NHID / 64, (N_NODES + 255) / 256), 256, 0, stream>>>(
      xb, W1T, s1, nullptr, N_NODES, NHID, NFEAT, 0);

  // 3. SpMM1: acc1[row] += val * s1[col]   (acc1 aliases xb)
  k_zero_f32<<<(unsigned)((nh1 + 255) / 256), 256, 0, stream>>>(acc1, nh1);
  k_spmm<<<(N_EDGES + EPB - 1) / EPB, NHID, 0, stream>>>(
      adj_row, adj_col, adj_val, s1, acc1, N_EDGES, NHID);

  // 4. h = bf16(relu(acc1 + b1))
  k_bias_relu_bf16<<<(unsigned)((nh1 + 255) / 256), 256, 0, stream>>>(
      acc1, b1, hb, nh1, NHID - 1);

  // 5. layer 2 GEMM: s2 = hb @ W2  (50000x256x128)  [s2 aliases s1]
  k_gemm_bf16<<<dim3(NCLASS / 64, (N_NODES + 255) / 256), 256, 0, stream>>>(
      hb, W2T, s2, nullptr, N_NODES, NCLASS, NHID, 0);

  // 6. SpMM2: acc2[row] += val * s2[col]   [acc2 aliases hb]
  k_zero_f32<<<(unsigned)((nh2 + 255) / 256), 256, 0, stream>>>(acc2, nh2);
  k_spmm<<<(N_EDGES + EPB - 1) / EPB, NCLASS, 0, stream>>>(
      adj_row, adj_col, adj_val, s2, acc2, N_EDGES, NCLASS);

  // 7. gathers (+b2): zu_in = bf16(z[u]); zvb = bf16(z[v])
  k_gather_bias_bf16<<<N_UV, NCLASS, 0, stream>>>(acc2, b2, u, zuin, NCLASS);
  k_gather_bias_bf16<<<N_UV, NCLASS, 0, stream>>>(acc2, b2, v, zvb,  NCLASS);

  // 8. zu = zu_in @ We^T  (2048x128x128): BT = We itself; bf16 output
  k_gemm_bf16<<<dim3(NCLASS / 64, N_UV / 256), 256, 0, stream>>>(
      zuin, Web, nullptr, zub, N_UV, NCLASS, NCLASS, 2);

  // 9. p_uv = sigmoid(zu @ zv^T): BT = zvb; fused sigmoid -> d_out
  k_gemm_bf16<<<dim3(N_UV / 64, N_UV / 256), 256, 0, stream>>>(
      zub, zvb, out, nullptr, N_UV, N_UV, NCLASS, 1);
}
